// KeypointPatchEmbed_60430189855075
// MI455X (gfx1250) — matrix operand order repaired and checked by reading, not compiled, not run on hardware
//
#include <hip/hip_runtime.h>
#include <hip/hip_bf16.h>
#include <stdint.h>

typedef __attribute__((ext_vector_type(16))) _Float16 v16h;
typedef __attribute__((ext_vector_type(8)))  float    v8f;

#define B_   8
#define T_   8
#define O_   25
#define C_   3
#define H_   192
#define W_   640
#define P_   16
#define D_   768
#define GH_  12
#define GW_  2
#define N_ROI (B_*T_*O_)      // 1600
#define K_    (C_*P_*P_)      // 768
#define HW_   (H_*W_)         // 122880

// ---------------------------------------------------------------------------
// Kernel 1: ROI-Align pooling.  One block (256 threads) per ROI -> gh/gw are
// uniform across each wave (no divergence).  Accumulate fp32, emit fp16 into
// pooled[N][K] with K ordered (i, j, c) to match einsum 'nijc'.
// ---------------------------------------------------------------------------
__global__ __launch_bounds__(256) void roi_pool_kernel(
    const float* __restrict__ features,   // (B, C, T, H, W)
    const float* __restrict__ kpts,       // (N_ROI, 3)
    _Float16*    __restrict__ pooled)     // (N_ROI, K_)
{
    const int n = blockIdx.x;             // ROI index
    const int p = threadIdx.x;            // bin index 0..255
    const int i = p >> 4;                 // patch row
    const int j = p & 15;                 // patch col

    const float kx = kpts[3*n + 0];
    const float ky = kpts[3*n + 1];
    const float kr = kpts[3*n + 2];

    const float bx1 = fmaxf(kx - kr, 0.0f);
    const float by1 = fmaxf(ky - kr, 0.0f);
    const float bx2 = fminf(kx + kr, (float)W_);
    const float by2 = fminf(kx + kr, (float)H_);   // reference quirk: uses kx

    const float start_h = by1 - 0.5f;
    const float start_w = bx1 - 0.5f;
    const float roi_h   = by2 - by1;
    const float roi_w   = bx2 - bx1;
    const float bin_h   = roi_h * (1.0f / P_);
    const float bin_w   = roi_w * (1.0f / P_);

    int gh = (int)ceilf(roi_h * (1.0f / P_));
    int gw = (int)ceilf(roi_w * (1.0f / P_));
    const float count = (float)max(gh * gw, 1);    // count uses unclamped grid
    if (gh > GH_) gh = GH_;
    if (gw > GW_) gw = GW_;
    const float ghf = (float)max(gh, 1);
    const float gwf = (float)max(gw, 1);

    const int img = n / O_;               // 0..63
    const int b   = img / T_;
    const int t   = img % T_;
    const float* __restrict__ f0 = features + (size_t)((b*C_ + 0)*T_ + t) * HW_;
    const float* __restrict__ f1 = features + (size_t)((b*C_ + 1)*T_ + t) * HW_;
    const float* __restrict__ f2 = features + (size_t)((b*C_ + 2)*T_ + t) * HW_;

    float acc0 = 0.0f, acc1 = 0.0f, acc2 = 0.0f;

    for (int iy = 0; iy < gh; ++iy) {
        const float cy = start_h + (float)i * bin_h + ((float)iy + 0.5f) * (bin_h / ghf);
        const bool  vy = (cy >= -1.0f) && (cy <= (float)H_);
        const float ccy = fminf(fmaxf(cy, 0.0f), (float)(H_ - 1));
        const int   y0  = (int)floorf(ccy);
        const int   y1  = min(y0 + 1, H_ - 1);
        const float ly  = ccy - (float)y0;
        const float hy  = 1.0f - ly;
        for (int ix = 0; ix < gw; ++ix) {
            const float cx = start_w + (float)j * bin_w + ((float)ix + 0.5f) * (bin_w / gwf);
            const bool  vx = (cx >= -1.0f) && (cx <= (float)W_);
            if (!(vy && vx)) continue;
            const float ccx = fminf(fmaxf(cx, 0.0f), (float)(W_ - 1));
            const int   x0  = (int)floorf(ccx);
            const int   x1  = min(x0 + 1, W_ - 1);
            const float lx  = ccx - (float)x0;
            const float hx  = 1.0f - lx;

            const float w00 = hy*hx, w01 = hy*lx, w10 = ly*hx, w11 = ly*lx;
            const int i00 = y0*W_ + x0, i01 = y0*W_ + x1;
            const int i10 = y1*W_ + x0, i11 = y1*W_ + x1;

            acc0 += w00*f0[i00] + w01*f0[i01] + w10*f0[i10] + w11*f0[i11];
            acc1 += w00*f1[i00] + w01*f1[i01] + w10*f1[i10] + w11*f1[i11];
            acc2 += w00*f2[i00] + w01*f2[i01] + w10*f2[i10] + w11*f2[i11];
        }
    }

    const float inv = 1.0f / count;
    const int base = n * K_ + (i * P_ + j) * C_;
    pooled[base + 0] = (_Float16)(acc0 * inv);
    pooled[base + 1] = (_Float16)(acc1 * inv);
    pooled[base + 2] = (_Float16)(acc2 * inv);
}

// ---------------------------------------------------------------------------
// Kernel 2: conv_w (D, C, P, P) fp32  ->  Bt[d][k] fp16, k = (i*P + j)*C + c.
// Row-major D x K so that WMMA B-fragment dword pairs {B[k][n], B[k+1][n]}
// are contiguous in memory.
// ---------------------------------------------------------------------------
__global__ __launch_bounds__(256) void weight_transform_kernel(
    const float* __restrict__ conv_w,
    _Float16*    __restrict__ bt)
{
    const int tid = blockIdx.x * 256 + threadIdx.x;
    if (tid >= D_ * K_) return;
    const int d  = tid / K_;
    const int k  = tid % K_;
    const int c  = k % C_;
    const int ij = k / C_;
    const int i  = ij / P_;
    const int j  = ij % P_;
    bt[tid] = (_Float16)conv_w[((d*C_ + c)*P_ + i)*P_ + j];
}

// ---------------------------------------------------------------------------
// Kernel 3: WMMA GEMM  out[M=1600][D=768] = pooled[M][K] * Bt[D][K]^T + bias.
// One wave -> 32x64 output tile: 2 M-subtiles x 4 N-subtiles = 8 f32
// accumulators.  Each B fragment is reused by 2 WMMAs (A0, A1), so the
// scheduler has independent WMMA work to overlap the next B-tile loads with.
// K consumed in 24 steps of 32 via v_wmma_f32_16x16x32_f16.
// Fragment layouts per CDNA5 ISA 7.12.2.
// ---------------------------------------------------------------------------
__global__ __launch_bounds__(256) void wmma_gemm_kernel(
    const _Float16* __restrict__ pooled,   // M x K
    const _Float16* __restrict__ bt,       // D x K
    const float*    __restrict__ bias,     // D
    float*          __restrict__ out)      // M x D
{
    const int lane = threadIdx.x & 31;
    const int wave = threadIdx.x >> 5;
    const int wgid = blockIdx.x * 8 + wave;     // 0..599
    const int mt   = wgid / 12;                 // 32-row M tile, 0..49
    const int ng   = wgid % 12;                 // 64-col N group

    const int m = lane & 15;                    // A row / C-D column
    const int h = lane >> 4;                    // half-wave select

    // A fragment source rows (dword view of fp16 pairs)
    const uint32_t* __restrict__ arow0 =
        (const uint32_t*)(pooled + (size_t)(mt * 32 + m) * K_);
    const uint32_t* __restrict__ arow1 =
        (const uint32_t*)(pooled + (size_t)(mt * 32 + 16 + m) * K_);

    // B fragment source rows: output column (ng*4+t)*16 + m maps to Bt row
    const uint32_t* __restrict__ brow[4];
#pragma unroll
    for (int t = 0; t < 4; ++t)
        brow[t] = (const uint32_t*)(bt + (size_t)((ng*4 + t)*16 + m) * K_);

    v8f acc[4][2];
#pragma unroll
    for (int t = 0; t < 4; ++t) {
        acc[t][0] = (v8f){};
        acc[t][1] = (v8f){};
    }

    for (int k0 = 0; k0 < K_; k0 += 32) {
        // --- A fragments (16x32 f16): VGPR v holds K pair:
        //     v<4 : K = 2v + 8h ; v>=4 : K = 2(v-4) + 8h + 16
        union { uint32_t u[8]; v16h v; } a0, a1;
#pragma unroll
        for (int v = 0; v < 8; ++v) {
            const int kk = (v < 4) ? (2*v + 8*h) : (2*(v - 4) + 8*h + 16);
            a0.u[v] = arow0[(k0 + kk) >> 1];
            a1.u[v] = arow1[(k0 + kk) >> 1];
        }

        // --- B fragments (32x16 f16): VGPR v holds K pair (same as A) for col m
#pragma unroll
        for (int t = 0; t < 4; ++t) {
            union { uint32_t u[8]; v16h v; } b;
#pragma unroll
            for (int v = 0; v < 8; ++v)
                b.u[v] = brow[t][(k0 + ((v < 4) ? (2*v + 8*h) : (2*(v - 4) + 8*h + 16))) >> 1];

            acc[t][0] = __builtin_amdgcn_wmma_f32_16x16x32_f16(
                false, a0.v, false, b.v, (short)0, acc[t][0], false, false);
            acc[t][1] = __builtin_amdgcn_wmma_f32_16x16x32_f16(
                false, a1.v, false, b.v, (short)0, acc[t][1], false, false);
        }
    }

    // --- Store D fragments: col = m (+ tile base), row = v + 8h (+ subtile)
#pragma unroll
    for (int t = 0; t < 4; ++t) {
        const int dcol = (ng*4 + t)*16 + m;
        const float bv = bias[dcol];
#pragma unroll
        for (int r = 0; r < 2; ++r) {
#pragma unroll
            for (int v = 0; v < 8; ++v) {
                const int row = mt*32 + r*16 + v + 8*h;
                out[(size_t)row * D_ + dcol] = acc[t][r][v] + bv;
            }
        }
    }
}

// ---------------------------------------------------------------------------
extern "C" void kernel_launch(void* const* d_in, const int* in_sizes, int n_in,
                              void* d_out, int out_size, void* d_ws, size_t ws_size,
                              hipStream_t stream) {
    const float* features  = (const float*)d_in[0];
    const float* keypoints = (const float*)d_in[1];
    const float* conv_w    = (const float*)d_in[2];
    const float* conv_b    = (const float*)d_in[3];
    float* out = (float*)d_out;

    _Float16* pooled = (_Float16*)d_ws;                                   // 2,457,600 B
    _Float16* bt     = (_Float16*)((char*)d_ws + (size_t)N_ROI * K_ * 2); // 1,179,648 B

    // 1) ROI-Align pooling: one block per ROI
    roi_pool_kernel<<<N_ROI, 256, 0, stream>>>(features, keypoints, pooled);

    // 2) Weight transform: D_*K_ = 589,824 elements
    weight_transform_kernel<<<(D_ * K_ + 255) / 256, 256, 0, stream>>>(conv_w, bt);

    // 3) WMMA GEMM: 50 M-tiles * 12 N-groups = 600 waves = 75 blocks of 8 waves
    wmma_gemm_kernel<<<75, 256, 0, stream>>>(pooled, bt, conv_b, out);
}
